// BiLSTM_CRF_58798102282586
// MI455X (gfx1250) — compile-verified, matrix operand order checked by reading, not run on hardware
//
#include <hip/hip_runtime.h>
#include <hip/hip_bf16.h>

typedef __attribute__((ext_vector_type(16))) _Float16 v16h;
typedef __attribute__((ext_vector_type(8)))  float    v8f;

#define S_LEN 512
#define BATCH 64
#define EMB   512
#define HID   512
#define HH    256
#define G4    1024   // 4*HH
#define TAGS  20
#define TP    32     // padded tag count
#define NEGV  -10000.0f

__device__ __forceinline__ float sigmoidf_(float x) {
    return 1.0f / (1.0f + __expf(-x));
}

// ---------------------------------------------------------------------------
// K1: embedding gather + fp32 -> fp16
// ---------------------------------------------------------------------------
__global__ void gather_embed_k(const int* __restrict__ sentence,
                               const float* __restrict__ emb,
                               _Float16* __restrict__ X16) {
    int row = blockIdx.x;                       // 0 .. S*B-1  (row = s*B + b)
    int tok = sentence[row];
    const float* src = emb + (size_t)tok * EMB;
    _Float16* dst = X16 + (size_t)row * EMB;
    for (int k = threadIdx.x; k < EMB; k += blockDim.x)
        dst[k] = (_Float16)src[k];
}

// ---------------------------------------------------------------------------
// K2: convert weights to f16, fuse biases
// ---------------------------------------------------------------------------
__global__ void convert_weights_k(const float* __restrict__ Wih_f, const float* __restrict__ Wih_b,
                                  const float* __restrict__ Whh_f, const float* __restrict__ Whh_b,
                                  const float* __restrict__ W_out,
                                  const float* __restrict__ bih_f, const float* __restrict__ bhh_f,
                                  const float* __restrict__ bih_b, const float* __restrict__ bhh_b,
                                  const float* __restrict__ b_out,
                                  _Float16* __restrict__ Wih16, _Float16* __restrict__ Whh16,
                                  _Float16* __restrict__ Wout16,
                                  float* __restrict__ biasG, float* __restrict__ boutP) {
    const int nWih  = 2 * G4 * EMB;   // 1048576
    const int nWhh  = 2 * G4 * HH;    // 524288
    const int nWout = TP * HID;       // 16384
    const int nBias = 2 * G4;         // 2048
    const int nBout = TP;             // 32
    int i = blockIdx.x * blockDim.x + threadIdx.x;
    if (i < nWih) {
        int d = i / (G4 * EMB), r = i % (G4 * EMB);
        Wih16[i] = (_Float16)(d ? Wih_b[r] : Wih_f[r]);
        return;
    }
    i -= nWih;
    if (i < nWhh) {
        int d = i / (G4 * HH), r = i % (G4 * HH);
        Whh16[i] = (_Float16)(d ? Whh_b[r] : Whh_f[r]);
        return;
    }
    i -= nWhh;
    if (i < nWout) {
        int n = i / HID, k = i % HID;
        Wout16[i] = (_Float16)((n < TAGS) ? W_out[n * HID + k] : 0.0f);
        return;
    }
    i -= nWout;
    if (i < nBias) {
        int d = i / G4, n = i % G4;
        biasG[i] = d ? (bih_b[n] + bhh_b[n]) : (bih_f[n] + bhh_f[n]);
        return;
    }
    i -= nBias;
    if (i < nBout) {
        boutP[i] = (i < TAGS) ? b_out[i] : NEGV;
    }
}

// ---------------------------------------------------------------------------
// K3: input projection GEMM, Xproj[dir][r][n] = X @ Wih^T + (bih+bhh)
//     M = S*B = 32768, N = 1024/dir, K = 512.
//     Register-blocked 32x32 supertile per wave (2 A-frag x 2 B-frag x 4 acc)
//     => halves both X and W HBM traffic vs 16x16/wave.
// ---------------------------------------------------------------------------
__global__ void xproj_gemm_k(const _Float16* __restrict__ X16,
                             const _Float16* __restrict__ Wih16,
                             const float* __restrict__ biasG,
                             float* __restrict__ Xproj) {
    int gw   = (blockIdx.x * blockDim.x + threadIdx.x) >> 5;  // global wave id
    int lane = threadIdx.x & 31;
    int ntAll = gw & 63;          // 64 N-supertiles (2 dirs x 32)
    int mt2   = gw >> 6;          // 0 .. 1023   (M supertile of 32 rows)
    int dir   = ntAll >> 5;
    int nt2   = ntAll & 31;       // N supertile of 32 cols
    int l15 = lane & 15, lhi = lane >> 4;
    int m0 = mt2 * 32, n0 = nt2 * 32;

    const _Float16* Ap0 = X16 + (size_t)(m0 + l15) * EMB + lhi * 16;
    const _Float16* Ap1 = Ap0 + (size_t)16 * EMB;
    const _Float16* Bp0 = Wih16 + (size_t)dir * G4 * EMB + (size_t)(n0 + l15) * EMB + lhi * 16;
    const _Float16* Bp1 = Bp0 + (size_t)16 * EMB;

    float bv0 = biasG[dir * G4 + n0 + l15];
    float bv1 = biasG[dir * G4 + n0 + 16 + l15];
    v8f acc00 = {bv0, bv0, bv0, bv0, bv0, bv0, bv0, bv0};
    v8f acc01 = {bv1, bv1, bv1, bv1, bv1, bv1, bv1, bv1};
    v8f acc10 = acc00, acc11 = acc01;

#pragma unroll
    for (int kt = 0; kt < 16; ++kt) {
        v16h a0 = *(const v16h*)(Ap0 + kt * 32);
        v16h a1 = *(const v16h*)(Ap1 + kt * 32);
        v16h b0 = *(const v16h*)(Bp0 + kt * 32);
        v16h b1 = *(const v16h*)(Bp1 + kt * 32);
        acc00 = __builtin_amdgcn_wmma_f32_16x16x32_f16(false, a0, false, b0, (short)0, acc00, false, false);
        acc01 = __builtin_amdgcn_wmma_f32_16x16x32_f16(false, a0, false, b1, (short)0, acc01, false, false);
        acc10 = __builtin_amdgcn_wmma_f32_16x16x32_f16(false, a1, false, b0, (short)0, acc10, false, false);
        acc11 = __builtin_amdgcn_wmma_f32_16x16x32_f16(false, a1, false, b1, (short)0, acc11, false, false);
    }

    float* Dbase = Xproj + (size_t)dir * S_LEN * BATCH * G4;
    float* D00 = Dbase + (size_t)(m0 + lhi * 8) * G4 + n0 + l15;
#pragma unroll
    for (int r = 0; r < 8; ++r) {
        D00[(size_t)r * G4]                       = acc00[r];
        D00[(size_t)r * G4 + 16]                  = acc01[r];
        D00[((size_t)r + 16) * G4]                = acc10[r];
        D00[((size_t)r + 16) * G4 + 16]           = acc11[r];
    }
}

// ---------------------------------------------------------------------------
// K4: recurrent LSTM. One block per direction, 1024 threads = 32 waves.
//     Dynamic LDS: gbuf f32[64][1024] (256KB) + hbuf f16[64][256] (32KB).
//     Per step: g = Xproj[t] + h @ Whh^T (WMMA), barrier, cell update, barrier.
//     Prefetch next step's Xproj C-init rows to hide the barrier-open latency.
// ---------------------------------------------------------------------------
__global__ void lstm_recurrent_k(const _Float16* __restrict__ Whh16,
                                 const float* __restrict__ Xproj,
                                 const float* __restrict__ h0,
                                 const float* __restrict__ c0,
                                 _Float16* __restrict__ lout) {
    extern __shared__ char smem[];
    float*    gbuf = (float*)smem;                               // [64][1024]
    _Float16* hbuf = (_Float16*)(smem + (size_t)BATCH * G4 * 4); // [64][256]

    const int dir  = blockIdx.x;
    const int tid  = threadIdx.x;
    const int wv   = tid >> 5;
    const int lane = tid & 31;
    const int l15 = lane & 15, lhi = lane >> 4;

    // --- init h, c: thread owns (m = tid/16, u in [ub, ub+16)) ---
    const int m  = tid >> 4;
    const int ub = (tid & 15) * 16;
    float creg[16];
#pragma unroll
    for (int j = 0; j < 16; ++j) {
        creg[j] = c0[(size_t)dir * BATCH * HH + m * HH + ub + j];
        hbuf[m * HH + ub + j] = (_Float16)h0[(size_t)dir * BATCH * HH + m * HH + ub + j];
    }
    __syncthreads();

    const int mtile = wv >> 3;          // 0..3 (M tile of 16 rows)
    const int ntb   = (wv & 7) * 8;     // base of 8 N tiles
    const float*    xp = Xproj + (size_t)dir * S_LEN * BATCH * G4;
    const _Float16* Wd = Whh16 + (size_t)dir * G4 * HH;

    for (int step = 0; step < S_LEN; ++step) {
        const int t = dir ? (S_LEN - 1 - step) : step;

        // prefetch next step's C-init rows (global_prefetch_b8)
        if (step + 1 < S_LEN) {
            const int tn = dir ? (S_LEN - 2 - step) : (step + 1);
            const float* pf = xp + ((size_t)tn * BATCH + mtile * 16 + lhi * 8) * G4
                                 + ntb * 16 + l15;
            __builtin_prefetch(pf, 0, 1);
        }

        // A fragments: h tile rows mtile*16.., all 8 K-tiles (K = 256)
        v16h a[8];
#pragma unroll
        for (int kt = 0; kt < 8; ++kt)
            a[kt] = *(const v16h*)(hbuf + (mtile * 16 + l15) * HH + kt * 32 + lhi * 16);

#pragma unroll
        for (int nti = 0; nti < 8; ++nti) {
            const int nt   = ntb + nti;
            const int ncol = nt * 16 + l15;
            // C init = input projection (bias already fused)
            const float* cp = xp + ((size_t)t * BATCH + mtile * 16 + lhi * 8) * G4 + ncol;
            v8f acc;
#pragma unroll
            for (int r = 0; r < 8; ++r) acc[r] = cp[(size_t)r * G4];
            const _Float16* bp = Wd + (size_t)ncol * HH + lhi * 16;
#pragma unroll
            for (int kt = 0; kt < 8; ++kt) {
                v16h b = *(const v16h*)(bp + kt * 32);
                acc = __builtin_amdgcn_wmma_f32_16x16x32_f16(false, a[kt], false, b,
                                                             (short)0, acc, false, false);
            }
#pragma unroll
            for (int r = 0; r < 8; ++r)
                gbuf[(mtile * 16 + lhi * 8 + r) * G4 + ncol] = acc[r];
        }
        __syncthreads();

        // --- elementwise LSTM cell: gates i,f,g,o are 256-wide chunks ---
        const float* gm = gbuf + m * G4;
#pragma unroll
        for (int j = 0; j < 16; ++j) {
            const int u = ub + j;
            float gi = sigmoidf_(gm[u]);
            float gf = sigmoidf_(gm[HH + u]);
            float gg = tanhf(gm[2 * HH + u]);
            float go = sigmoidf_(gm[3 * HH + u]);
            creg[j] = gf * creg[j] + gi * gg;
            float hv = go * tanhf(creg[j]);
            hbuf[m * HH + u] = (_Float16)hv;
            lout[((size_t)t * BATCH + m) * HID + dir * HH + u] = (_Float16)hv;
        }
        __syncthreads();
    }
}

// ---------------------------------------------------------------------------
// K5: feats GEMM, [32768 x 512] x [512 x 32] + b_out (padded).
//     One wave computes a 16x32 strip: A fragment shared across both N-tiles.
// ---------------------------------------------------------------------------
__global__ void feats_gemm_k(const _Float16* __restrict__ lout,
                             const _Float16* __restrict__ Wout16,
                             const float* __restrict__ boutP,
                             float* __restrict__ feats) {
    int gw   = (blockIdx.x * blockDim.x + threadIdx.x) >> 5;  // M tile id, 0..2047
    int lane = threadIdx.x & 31;
    int l15 = lane & 15, lhi = lane >> 4;

    const _Float16* Ap  = lout + (size_t)(gw * 16 + l15) * HID + lhi * 16;
    const _Float16* Bp0 = Wout16 + (size_t)l15 * HID + lhi * 16;
    const _Float16* Bp1 = Bp0 + (size_t)16 * HID;
    float bv0 = boutP[l15];
    float bv1 = boutP[16 + l15];
    v8f acc0 = {bv0, bv0, bv0, bv0, bv0, bv0, bv0, bv0};
    v8f acc1 = {bv1, bv1, bv1, bv1, bv1, bv1, bv1, bv1};
#pragma unroll
    for (int kt = 0; kt < 16; ++kt) {
        v16h a  = *(const v16h*)(Ap + kt * 32);
        v16h b0 = *(const v16h*)(Bp0 + kt * 32);
        v16h b1 = *(const v16h*)(Bp1 + kt * 32);
        acc0 = __builtin_amdgcn_wmma_f32_16x16x32_f16(false, a, false, b0, (short)0, acc0, false, false);
        acc1 = __builtin_amdgcn_wmma_f32_16x16x32_f16(false, a, false, b1, (short)0, acc1, false, false);
    }
    float* Dp = feats + (size_t)(gw * 16 + lhi * 8) * TP + l15;
#pragma unroll
    for (int r = 0; r < 8; ++r) {
        Dp[(size_t)r * TP]      = acc0[r];
        Dp[(size_t)r * TP + 16] = acc1[r];
    }
}

// ---------------------------------------------------------------------------
// K6: Viterbi decode + backtrack. One wave32 per batch element; lane = tag.
// ---------------------------------------------------------------------------
__global__ void viterbi_k(const float* __restrict__ feats,
                          const float* __restrict__ trans,
                          float* __restrict__ out) {
    __shared__ float st[TAGS * TAGS];
    __shared__ unsigned char bps[S_LEN * TAGS];

    const int b    = blockIdx.x;
    const int lane = threadIdx.x;
    for (int i = lane; i < TAGS * TAGS; i += 32) st[i] = trans[i];
    __syncthreads();

    const int next = lane;
    float fv = (next == 0) ? 0.0f : NEGV;   // START = 0

    for (int s = 0; s < S_LEN; ++s) {
        float best = -3.0e38f;
        int   barg = 0;
#pragma unroll
        for (int prev = 0; prev < TAGS; ++prev) {
            float fvp = __shfl(fv, prev, 32);
            float v = fvp + ((next < TAGS) ? st[next * TAGS + prev] : 0.0f);
            if (v > best) { best = v; barg = prev; }
        }
        float ft = (next < TAGS) ? feats[((size_t)s * BATCH + b) * TP + next] : 0.0f;
        fv = best + ft;
        if (next < TAGS) bps[s * TAGS + next] = (unsigned char)barg;
    }
    __syncthreads();

    // terminal = fv + trans[STOP=1][tag]; wave max + argmax
    float tv = (next < TAGS) ? (fv + st[1 * TAGS + next]) : -3.0e38f;
    int   ti = next;
#pragma unroll
    for (int off = 16; off > 0; off >>= 1) {
        float ov = __shfl_xor(tv, off, 32);
        int   oi = __shfl_xor(ti, off, 32);
        if (ov > tv || (ov == tv && oi < ti)) { tv = ov; ti = oi; }
    }

    if (lane == 0) {
        out[b] = tv;                                   // path_score [B,1] flat
        float* path = out + BATCH + (size_t)b * S_LEN; // best_path [B,S]
        int tag = ti;
        path[S_LEN - 1] = (float)tag;
        for (int s = S_LEN - 1; s >= 1; --s) {
            tag = bps[s * TAGS + tag];
            path[s - 1] = (float)tag;
        }
    }
}

// ---------------------------------------------------------------------------
// Launch
// ---------------------------------------------------------------------------
extern "C" void kernel_launch(void* const* d_in, const int* in_sizes, int n_in,
                              void* d_out, int out_size, void* d_ws, size_t ws_size,
                              hipStream_t stream) {
    const int*   sentence = (const int*)d_in[0];
    const float* emb   = (const float*)d_in[1];
    const float* Wih_f = (const float*)d_in[2];
    const float* Whh_f = (const float*)d_in[3];
    const float* bih_f = (const float*)d_in[4];
    const float* bhh_f = (const float*)d_in[5];
    const float* Wih_b = (const float*)d_in[6];
    const float* Whh_b = (const float*)d_in[7];
    const float* bih_b = (const float*)d_in[8];
    const float* bhh_b = (const float*)d_in[9];
    const float* h0    = (const float*)d_in[10];
    const float* c0    = (const float*)d_in[11];
    const float* W_out = (const float*)d_in[12];
    const float* b_out = (const float*)d_in[13];
    const float* trans = (const float*)d_in[14];
    float* out = (float*)d_out;

    char* ws = (char*)d_ws;
    size_t off = 0;
    auto take = [&](size_t bytes) { size_t o = off; off = (off + bytes + 255) & ~(size_t)255; return o; };

    _Float16* X16    = (_Float16*)(ws + take((size_t)S_LEN * BATCH * EMB * 2)); // 32 MB
    _Float16* Wih16  = (_Float16*)(ws + take((size_t)2 * G4 * EMB * 2));        // 2 MB
    _Float16* Whh16  = (_Float16*)(ws + take((size_t)2 * G4 * HH * 2));         // 1 MB
    _Float16* Wout16 = (_Float16*)(ws + take((size_t)TP * HID * 2));            // 32 KB
    float*    biasG  = (float*)(ws + take((size_t)2 * G4 * 4));                 // 8 KB
    float*    boutP  = (float*)(ws + take((size_t)TP * 4));
    float*    Xproj  = (float*)(ws + take((size_t)2 * S_LEN * BATCH * G4 * 4)); // 256 MB
    _Float16* lout   = (_Float16*)(ws + take((size_t)S_LEN * BATCH * HID * 2)); // 32 MB
    float*    feats  = (float*)(ws + take((size_t)S_LEN * BATCH * TP * 4));     // 4 MB
    (void)ws_size; (void)in_sizes; (void)n_in; (void)out_size;

    // K1: embedding gather -> f16
    gather_embed_k<<<S_LEN * BATCH, 256, 0, stream>>>(sentence, emb, X16);

    // K2: weight conversion
    {
        const int total = 2 * G4 * EMB + 2 * G4 * HH + TP * HID + 2 * G4 + TP;
        convert_weights_k<<<(total + 255) / 256, 256, 0, stream>>>(
            Wih_f, Wih_b, Whh_f, Whh_b, W_out,
            bih_f, bhh_f, bih_b, bhh_b, b_out,
            Wih16, Whh16, Wout16, biasG, boutP);
    }

    // K3: input projections, 1024 M-supertiles x 64 N-supertiles, 8 waves/block
    xproj_gemm_k<<<8192, 256, 0, stream>>>(X16, Wih16, biasG, Xproj);

    // K4: recurrent BiLSTM — one workgroup per direction, 288 KB dynamic LDS
    {
        const size_t smem = (size_t)BATCH * G4 * 4 + (size_t)BATCH * HH * 2; // 294912
        lstm_recurrent_k<<<2, 1024, smem, stream>>>(Whh16, Xproj, h0, c0, lout);
    }

    // K5: emission features, 2048 waves (one 16x32 strip each)
    feats_gemm_k<<<256, 256, 0, stream>>>(lout, Wout16, boutP, feats);

    // K6: Viterbi + backtrack
    viterbi_k<<<BATCH, 32, 0, stream>>>(feats, trans, out);
}
